// Label_Attn_Fusion_Layer_for_token_16621523435984
// MI455X (gfx1250) — compile-verified, matrix operand order checked by reading, not run on hardware
//
#include <hip/hip_runtime.h>

// ---------------------------------------------------------------------------
// Label-attention fusion for MI455X (gfx1250, wave32, WMMA 16x16x32 bf16).
//
// Sizes: B=8 S=128 L=16 Ls=8 H=768 heads=12 hd=64 T=136 (pad 144)
// Unique projected rows: 8*128 token rows + 16*8 label rows = 1152.
// ---------------------------------------------------------------------------

#define H_DIM    768
#define N_HEADS  12
#define HEAD_DIM 64
#define BATCH    8
#define SEQ      128
#define LABELS   16
#define LSEQ     8
#define T_REAL   136          // SEQ + LSEQ
#define T_PAD    144          // next multiple of 16
#define UROWS    1152         // BATCH*SEQ + LABELS*LSEQ
#define WELEMS   (H_DIM * H_DIM)     // 589824
#define XELEMS   (UROWS * H_DIM)     // 884736

typedef __attribute__((ext_vector_type(16))) __bf16 v16bf;
typedef __attribute__((ext_vector_type(8)))  float  v8f;

union FragU { uint4 q[2]; v16bf v; };

// Load a 16x32 bf16 WMMA fragment. `rowbase` = per-lane pointer to element
// k=0 of this lane's row (A) / column (B); must be 16B aligned.
// Lane layout (ISA 7.12.2, 16-bit A 16x32): lanes 0-15 hold K {0..7,16..23},
// lanes 16-31 hold K {8..15,24..31} of row (lane&15). B uses the mirrored
// layout with "row" = column of B (i.e. a row of B^T).
static __device__ __forceinline__ v16bf frag_from_rowptr(const __bf16* rowbase) {
    const int lane = threadIdx.x & 31;
    const __bf16* p = rowbase + ((lane & 16) ? 8 : 0);
    FragU u;
    u.q[0] = *reinterpret_cast<const uint4*>(p);
    u.q[1] = *reinterpret_cast<const uint4*>(p + 16);
    return u.v;
}

// Fragment whose upper 16 K-values are zero (for the tail K-step of 16).
static __device__ __forceinline__ v16bf frag_lo16(const __bf16* rowbase) {
    const int lane = threadIdx.x & 31;
    const __bf16* p = rowbase + ((lane & 16) ? 8 : 0);
    FragU u;
    u.q[0] = *reinterpret_cast<const uint4*>(p);
    u.q[1] = make_uint4(0u, 0u, 0u, 0u);
    return u.v;
}

static __device__ __forceinline__ v8f wmma_bf16(v16bf a, v16bf b, v8f c) {
    // (neg_a, A, neg_b, B, c_mod, C, reuse_a, reuse_b)
    return __builtin_amdgcn_wmma_f32_16x16x32_bf16(false, a, false, b,
                                                   (short)0, c, false, false);
}

// Unique-row index for logical sequence position t of pair (b, l).
static __device__ __forceinline__ int urow(int b, int l, int t) {
    return (t < SEQ) ? (b * SEQ + t) : (BATCH * SEQ + l * LSEQ + (t - SEQ));
}

// ---------------------------------------------------------------------------
// Kernel 1a: fp32 -> bf16 conversion of the three weight matrices.
// ---------------------------------------------------------------------------
__global__ __launch_bounds__(256) void cvt_weights_kernel(
    const float* __restrict__ Wq, const float* __restrict__ Wk,
    const float* __restrict__ Wv, __bf16* __restrict__ out)
{
    int i = blockIdx.x * blockDim.x + threadIdx.x;
    if (i < 3 * WELEMS) {
        const float* src = (i < WELEMS) ? Wq : ((i < 2 * WELEMS) ? Wk : Wv);
        out[i] = (__bf16)src[i % WELEMS];
    }
}

// ---------------------------------------------------------------------------
// Kernel 1b: gather the 1152 unique rows (tokens then labels) as bf16.
// ---------------------------------------------------------------------------
__global__ __launch_bounds__(256) void cvt_x_kernel(
    const float* __restrict__ tok, const float* __restrict__ lab,
    __bf16* __restrict__ X)
{
    int i = blockIdx.x * blockDim.x + threadIdx.x;
    if (i < XELEMS) {
        int r = i / H_DIM;
        int h = i - r * H_DIM;
        float v = (r < BATCH * SEQ) ? tok[r * H_DIM + h]
                                    : lab[(r - BATCH * SEQ) * H_DIM + h];
        X[i] = (__bf16)v;
    }
}

// ---------------------------------------------------------------------------
// Kernel 2: QKV projection on unique rows.  y = x W^T + b  (NT GEMM).
// grid = (Mtiles=9, Ntiles=12, z=3{q,k,v}), 256 threads = 8 waves,
// wave -> 16(M) x 64(N) tile, K = 768 in 24 steps of 32.
// Software-pipelined AND fully unrolled: each stage gets its own registers
// (no cur<-nxt copies) and per-step addresses become immediate offsets.
// Q is pre-scaled by 1/sqrt(HEAD_DIM).
// ---------------------------------------------------------------------------
__global__ __launch_bounds__(256) void qkv_gemm_kernel(
    const __bf16* __restrict__ X,      // [1152,768]
    const __bf16* __restrict__ W,      // [3][768,768] row-major (out, in)
    const float*  __restrict__ bq, const float* __restrict__ bk,
    const float*  __restrict__ bv,
    __bf16* __restrict__ Out)          // [3][1152,768]
{
    const int z    = blockIdx.z;
    const int wave = threadIdx.x >> 5;
    const int lane = threadIdx.x & 31;
    const int r    = lane & 15;
    const int m0   = blockIdx.x * 128 + wave * 16;
    const int n0   = blockIdx.y * 64;

    const __bf16* Wz   = W + (size_t)z * WELEMS;
    const float*  bias = (z == 0) ? bq : ((z == 1) ? bk : bv);

    const __bf16* arow = X + (size_t)(m0 + r) * H_DIM;          // A: row m0+r
    const __bf16* brow[4];
#pragma unroll
    for (int j = 0; j < 4; ++j)
        brow[j] = Wz + (size_t)(n0 + j * 16 + r) * H_DIM;       // B col = W row

    v8f acc[4] = {v8f{}, v8f{}, v8f{}, v8f{}};

    // ---- software-pipelined, fully unrolled K loop ------------------------
    v16bf a_cur = frag_from_rowptr(arow);
    v16bf b_cur[4];
#pragma unroll
    for (int j = 0; j < 4; ++j) b_cur[j] = frag_from_rowptr(brow[j]);

#pragma unroll
    for (int k = 32; k < H_DIM; k += 32) {
        __builtin_prefetch(arow + k + 32, 0, 3);
        v16bf a_nxt = frag_from_rowptr(arow + k);
        v16bf b_nxt[4];
#pragma unroll
        for (int j = 0; j < 4; ++j) b_nxt[j] = frag_from_rowptr(brow[j] + k);
#pragma unroll
        for (int j = 0; j < 4; ++j) acc[j] = wmma_bf16(a_cur, b_cur[j], acc[j]);
        a_cur = a_nxt;
#pragma unroll
        for (int j = 0; j < 4; ++j) b_cur[j] = b_nxt[j];
    }
#pragma unroll
    for (int j = 0; j < 4; ++j) acc[j] = wmma_bf16(a_cur, b_cur[j], acc[j]);

    // ---- epilogue: bias, Q-scale, bf16 store ------------------------------
    const float scale = (z == 0) ? 0.125f : 1.0f;   // fold 1/sqrt(64) into Q
    const int   hi8   = (lane & 16) ? 8 : 0;
    __bf16* Oz = Out + (size_t)z * XELEMS;
#pragma unroll
    for (int j = 0; j < 4; ++j) {
        int   c  = n0 + j * 16 + r;
        float bb = bias[c];
#pragma unroll
        for (int i = 0; i < 8; ++i) {
            int row = m0 + i + hi8;                 // C layout: VGPR i -> M=i/i+8
            Oz[(size_t)row * H_DIM + c] = (__bf16)((acc[j][i] + bb) * scale);
        }
    }
}

// ---------------------------------------------------------------------------
// Kernel 3: attention per (pair, head).  grid = (128, 12), 288 thr = 9 waves.
// Each wave owns one 16-row tile of the 144x144 (padded) score matrix:
//   S = Q K^T (Q pre-scaled) + mask-bias, softmax over 144 cols in registers,
//   P (bf16) -> LDS, ctx = P V via LDS-staged V^T.
// V is staged row-major with GLOBAL_LOAD_ASYNC_TO_LDS_B128 (ASYNCcnt), then
// transposed in LDS.  The staging buffer aliases the P buffer (its lifetime
// ends at the barrier before P is written).
// ---------------------------------------------------------------------------
__global__ __launch_bounds__(288) void attn_kernel(
    const __bf16* __restrict__ Qu, const __bf16* __restrict__ Ku,
    const __bf16* __restrict__ Vu,
    const float* __restrict__ token_mask,   // [B,S]
    const float* __restrict__ label_mask,   // [L,Ls]
    float* __restrict__ out)                // [B,T,L,H]
{
    __shared__ __align__(16) unsigned char smem[
        (T_PAD * T_PAD + HEAD_DIM * T_PAD) * sizeof(__bf16)];
    __bf16* Ps = reinterpret_cast<__bf16*>(smem);            // P : [144][144]
    __bf16* vT = Ps + T_PAD * T_PAD;                         // V^T: [64][144]
    __bf16* Vs = Ps;   // async staging alias: V row-major [136][64]

    const int bl   = blockIdx.x;
    const int head = blockIdx.y;
    const int b    = bl >> 4;
    const int l    = bl & 15;
    const int tid  = threadIdx.x;
    const int w    = tid >> 5;      // 0..8 : row tile
    const int lane = tid & 31;
    const int r    = lane & 15;
    const int hi8  = (lane & 16) ? 8 : 0;

    // ---- stage V row-major via async global->LDS b128 copies --------------
    {
        const int NB = (T_REAL * HEAD_DIM) / 8;   // 1088 x 16B transfers
        for (int i = tid; i < NB; i += 288) {
            int t  = i >> 3;                      // 8 groups of 8 bf16 per row
            int dg = (i & 7) * 8;
            const __bf16* g =
                Vu + (size_t)urow(b, l, t) * H_DIM + head * HEAD_DIM + dg;
            unsigned lds_off =
                (unsigned)(uintptr_t)(Vs + t * HEAD_DIM + dg);
            asm volatile("global_load_async_to_lds_b128 %0, %1, off"
                         :: "v"(lds_off), "v"(g) : "memory");
        }
        asm volatile("s_wait_asynccnt 0x0" ::: "memory");
    }
    __syncthreads();

    // ---- transpose to V^T (zero-pad t in [136,144)) -----------------------
    for (int idx = tid; idx < HEAD_DIM * T_PAD; idx += 288) {
        int d = idx / T_PAD;
        int t = idx - d * T_PAD;
        __bf16 v = (__bf16)0.0f;
        if (t < T_REAL) v = Vs[t * HEAD_DIM + d];
        vT[d * T_PAD + t] = v;
    }
    __syncthreads();   // Vs dead from here; Ps may be written after this

    // ---- scores: S[w*16 .. w*16+15][0..143] -------------------------------
    v8f sacc[9];
#pragma unroll
    for (int ct = 0; ct < 9; ++ct) sacc[ct] = v8f{};

    int tq = w * 16 + r; if (tq > T_REAL - 1) tq = T_REAL - 1;   // clamp pads
    const __bf16* qrow = Qu + (size_t)urow(b, l, tq) * H_DIM + head * HEAD_DIM;

#pragma unroll
    for (int ks = 0; ks < HEAD_DIM; ks += 32) {
        v16bf a = frag_from_rowptr(qrow + ks);
#pragma unroll
        for (int ct = 0; ct < 9; ++ct) {
            int tk = ct * 16 + r; if (tk > T_REAL - 1) tk = T_REAL - 1;
            const __bf16* krow =
                Ku + (size_t)urow(b, l, tk) * H_DIM + head * HEAD_DIM + ks;
            sacc[ct] = wmma_bf16(a, frag_from_rowptr(krow), sacc[ct]);
        }
    }

    // ---- additive mask bias ----------------------------------------------
#pragma unroll
    for (int ct = 0; ct < 9; ++ct) {
        int   c = ct * 16 + r;       // column index, same for both lane halves
        float bb;
        if (c < SEQ)         bb = (1.0f - token_mask[b * SEQ + c]) * -10000.0f;
        else if (c < T_REAL) bb = (1.0f - label_mask[l * LSEQ + (c - SEQ)]) * -10000.0f;
        else                 bb = -1.0e30f;   // pad columns -> prob 0
#pragma unroll
        for (int i = 0; i < 8; ++i) sacc[ct][i] += bb;
    }

    // ---- softmax over the 144 columns (rows live in lanes of one half) ----
    float mx[8], sm[8];
#pragma unroll
    for (int i = 0; i < 8; ++i) {
        float m = sacc[0][i];
#pragma unroll
        for (int ct = 1; ct < 9; ++ct) m = fmaxf(m, sacc[ct][i]);
        mx[i] = m;
    }
#pragma unroll
    for (int s = 1; s < 16; s <<= 1)
#pragma unroll
        for (int i = 0; i < 8; ++i)
            mx[i] = fmaxf(mx[i], __shfl_xor(mx[i], s, 32));

#pragma unroll
    for (int i = 0; i < 8; ++i) sm[i] = 0.0f;
#pragma unroll
    for (int ct = 0; ct < 9; ++ct)
#pragma unroll
        for (int i = 0; i < 8; ++i) {
            float e = __expf(sacc[ct][i] - mx[i]);
            sacc[ct][i] = e;
            sm[i] += e;
        }
#pragma unroll
    for (int s = 1; s < 16; s <<= 1)
#pragma unroll
        for (int i = 0; i < 8; ++i)
            sm[i] += __shfl_xor(sm[i], s, 32);

    float inv[8];
#pragma unroll
    for (int i = 0; i < 8; ++i) inv[i] = 1.0f / sm[i];

    // ---- P -> LDS (C-layout scatter; every row/col of [144][144] covered) --
#pragma unroll
    for (int ct = 0; ct < 9; ++ct)
#pragma unroll
        for (int i = 0; i < 8; ++i)
            Ps[(w * 16 + i + hi8) * T_PAD + ct * 16 + r] =
                (__bf16)(sacc[ct][i] * inv[i]);
    __syncthreads();

    // ---- ctx = P (16x144) * V (144x64): 4 full K-steps + one 16-wide tail --
    v8f cacc[4] = {v8f{}, v8f{}, v8f{}, v8f{}};
    const __bf16* prow = Ps + (size_t)(w * 16 + r) * T_PAD;

#pragma unroll
    for (int ks = 0; ks < 128; ks += 32) {
        v16bf a = frag_from_rowptr(prow + ks);
#pragma unroll
        for (int j = 0; j < 4; ++j)
            cacc[j] = wmma_bf16(a,
                frag_from_rowptr(vT + (size_t)(j * 16 + r) * T_PAD + ks),
                cacc[j]);
    }
    {
        v16bf a = frag_lo16(prow + 128);
#pragma unroll
        for (int j = 0; j < 4; ++j)
            cacc[j] = wmma_bf16(a,
                frag_lo16(vT + (size_t)(j * 16 + r) * T_PAD + 128),
                cacc[j]);
    }

    // ---- scatter to output layout [B, T, L, H] ----------------------------
#pragma unroll
    for (int j = 0; j < 4; ++j)
#pragma unroll
        for (int i = 0; i < 8; ++i) {
            int t = w * 16 + i + hi8;
            if (t < T_REAL) {
                size_t o = (((size_t)b * T_REAL + t) * LABELS + l) * H_DIM
                         + head * HEAD_DIM + j * 16 + r;
                out[o] = cacc[j][i];
            }
        }
}

// ---------------------------------------------------------------------------
// Host launcher (graph-capture safe: only kernel launches on `stream`).
// Workspace map (bf16 elements): W[3*589824] | X[884736] | Q | K | V.
// Total 5,308,416 bf16 = 10.6 MB.
// ---------------------------------------------------------------------------
extern "C" void kernel_launch(void* const* d_in, const int* in_sizes, int n_in,
                              void* d_out, int out_size, void* d_ws, size_t ws_size,
                              hipStream_t stream) {
    const float* tok   = (const float*)d_in[0];
    const float* lab   = (const float*)d_in[1];
    const float* tmask = (const float*)d_in[2];
    const float* lmask = (const float*)d_in[3];
    const float* Wq    = (const float*)d_in[4];
    const float* bq    = (const float*)d_in[5];
    const float* Wk    = (const float*)d_in[6];
    const float* bk    = (const float*)d_in[7];
    const float* Wv    = (const float*)d_in[8];
    const float* bv    = (const float*)d_in[9];

    __bf16* ws = (__bf16*)d_ws;
    __bf16* Wb = ws;                       // [3][768*768]
    __bf16* X  = Wb + 3 * (size_t)WELEMS;  // [1152*768]
    __bf16* Qu = X  + (size_t)XELEMS;      // [1152*768]
    __bf16* Ku = Qu + (size_t)XELEMS;
    __bf16* Vu = Ku + (size_t)XELEMS;
    // Note: qkv_gemm_kernel writes Out + z*XELEMS, so passing Qu covers Q/K/V.

    cvt_weights_kernel<<<(3 * WELEMS + 255) / 256, 256, 0, stream>>>(Wq, Wk, Wv, Wb);
    cvt_x_kernel<<<(XELEMS + 255) / 256, 256, 0, stream>>>(tok, lab, X);
    qkv_gemm_kernel<<<dim3(UROWS / 128, H_DIM / 64, 3), 256, 0, stream>>>(
        X, Wb, bq, bk, bv, Qu);
    attn_kernel<<<dim3(BATCH * LABELS, N_HEADS), 288, 0, stream>>>(
        Qu, Ku, Vu, tmask, lmask, (float*)d_out);
}
